// GATA_28329604285245
// MI455X (gfx1250) — matrix operand order repaired and checked by reading, not compiled, not run on hardware
//
#include <hip/hip_runtime.h>

#define NN 10000
#define EE 160000
#define DNE 128
#define HH 8
#define LSTR 136   // 128 + 8 halves padding (272B rows -> bank rotation)

typedef __attribute__((ext_vector_type(16))) _Float16 v16h;
typedef __attribute__((ext_vector_type(8)))  float    v8f;

union FragU { uint4 u[2]; v16h h; };
union H8   { _Float16 h[8]; uint4 u; };

// ---------- helpers ----------
__device__ __forceinline__ unsigned f2u(float f) {
    unsigned u = __float_as_uint(f);
    return (u & 0x80000000u) ? ~u : (u | 0x80000000u);
}
__device__ __forceinline__ float u2f(unsigned u) {
    unsigned v = (u & 0x80000000u) ? (u & 0x7fffffffu) : ~u;
    return __uint_as_float(v);
}
__device__ __forceinline__ float silu(float x) { return x / (1.0f + __expf(-x)); }

__device__ __forceinline__ void atomAddF(float* p, float v) {
    __hip_atomic_fetch_add(p, v, __ATOMIC_RELAXED, __HIP_MEMORY_SCOPE_AGENT);
}

// A fragment (16x32 f16) from a row-major LDS tile (stride LSTR halves).
// ISA layout: lane L holds row M=L&15; K(e) = kb + (e<8 ? e : e+8) + 8*(L>=16)
__device__ __forceinline__ v16h load_a(const _Float16* tile, int kb) {
    int lane = threadIdx.x & 31;
    int row  = lane & 15;
    int off  = kb + ((lane >> 4) << 3);
    const _Float16* p = tile + row * LSTR + off;
    FragU f;
    f.u[0] = *(const uint4*)(p);
    f.u[1] = *(const uint4*)(p + 16);
    return f.h;
}

// B fragment (32x16 f16) from pre-packed weights: tile = 512 halves,
// lane-major (lane*16 + e) so each lane does two contiguous b128 loads.
__device__ __forceinline__ v16h load_b(const _Float16* packed, int tileId) {
    const int lane = threadIdx.x & 31;
    const _Float16* p = packed + (size_t)tileId * 512 + lane * 16;
    FragU f;
    f.u[0] = *(const uint4*)(p);
    f.u[1] = *(const uint4*)(p + 8);
    return f.h;
}

__device__ __forceinline__ v8f wmma16(v16h a, v16h b, v8f c) {
    return __builtin_amdgcn_wmma_f32_16x16x32_f16(false, a, false, b, (short)0, c, false, false);
}

// ---------- pass 0: packing / conversion / init ----------
// Pack fp32 KxNcols weight into f16 B-fragment-tiled layout.
// tile id = kb*(ncols/16)+nt ; element: k = kb*32 + 8*(lane>>4) + (e<8 ? e : e+8) ; col = nt*16 + (lane&15)
__global__ void pack_b_kernel(const float* __restrict__ src, _Float16* __restrict__ dst, int ncols) {
    int idx = blockIdx.x * 256 + threadIdx.x;
    int total = 128 * ncols;
    if (idx >= total) return;
    int tile = idx >> 9, r = idx & 511, lane = r >> 4, e = r & 15;
    int ncols16 = ncols >> 4;
    int kb = tile / ncols16, nt = tile - kb * ncols16;
    int k   = kb * 32 + ((lane >> 4) << 3) + (e < 8 ? e : e + 8);
    int col = nt * 16 + (lane & 15);
    dst[idx] = (_Float16)src[(size_t)k * ncols + col];
}

__global__ void conv_h_kernel(const float* __restrict__ src, _Float16* __restrict__ dst, int total) {
    int idx = blockIdx.x * 256 + threadIdx.x;
    if (idx < total) dst[idx] = (_Float16)src[idx];
}

__global__ void copy_f_kernel(const float* __restrict__ src, float* __restrict__ dst, int total) {
    int idx = blockIdx.x * 256 + threadIdx.x;
    if (idx < total) dst[idx] = src[idx];
}

__global__ void init_stats_kernel(unsigned* mx, float* den, int* deg) {
    int idx = blockIdx.x * 256 + threadIdx.x;
    if (idx < NN * HH) { mx[idx] = 0u; den[idx] = 0.0f; }
    if (idx < NN) deg[idx] = 0;
}

// ---------- pass 1: alpha = sum_d q*(k*geom), segment max ----------
__global__ __launch_bounds__(256) void alpha_kernel(
    const _Float16* __restrict__ h16, const float* __restrict__ t,
    const int* __restrict__ edge,
    const _Float16* __restrict__ wq16, const _Float16* __restrict__ wk16,
    const _Float16* __restrict__ wre16,
    const float* __restrict__ bq, const float* __restrict__ bk, const float* __restrict__ bre,
    float* __restrict__ alpha, unsigned* __restrict__ mx)
{
    __shared__ _Float16 hiS[16 * LSTR], hjS[16 * LSTR], tS[16 * LSTR];
    __shared__ int is_[16], js_[16];
    const int tid = threadIdx.x;
    const int e0  = blockIdx.x * 16;

    if (tid < 16) {
        is_[tid] = edge[2 * (e0 + tid)];
        js_[tid] = edge[2 * (e0 + tid) + 1];
    }
    __syncthreads();

    // gather hi/hj rows (f16, 8 uint4 = 64 halves per row)
    if (tid < 128) {
        int row = tid >> 3, part = tid & 7;
        *(uint4*)(hiS + row * LSTR + part * 8) =
            ((const uint4*)(h16 + (size_t)is_[row] * DNE))[part];
    } else {
        int t2 = tid - 128, row = t2 >> 3, part = t2 & 7;
        *(uint4*)(hjS + row * LSTR + part * 8) =
            ((const uint4*)(h16 + (size_t)js_[row] * DNE))[part];
    }
    // convert t rows fp32 -> f16 (all 256 threads, 8 floats each)
    {
        int row = tid >> 4, part = tid & 15;
        const float* ts = t + (size_t)(e0 + row) * DNE + part * 8;
        float4 f0 = *(const float4*)(ts);
        float4 f1 = *(const float4*)(ts + 4);
        H8 hb;
        hb.h[0] = (_Float16)f0.x; hb.h[1] = (_Float16)f0.y;
        hb.h[2] = (_Float16)f0.z; hb.h[3] = (_Float16)f0.w;
        hb.h[4] = (_Float16)f1.x; hb.h[5] = (_Float16)f1.y;
        hb.h[6] = (_Float16)f1.z; hb.h[7] = (_Float16)f1.w;
        *(uint4*)(tS + row * LSTR + part * 8) = hb.u;
    }
    __syncthreads();

    const int wv = tid >> 5;        // wave == head
    const int lane = tid & 31;
    v8f cq = {}, ck = {}, cg = {};
    #pragma unroll
    for (int kb = 0; kb < 4; ++kb) {
        v16h ahi = load_a(hiS, kb * 32);
        v16h ahj = load_a(hjS, kb * 32);
        v16h at  = load_a(tS,  kb * 32);
        int tileId = kb * 8 + wv;
        v16h bq_ = load_b(wq16, tileId);
        v16h bk_ = load_b(wk16, tileId);
        v16h bg_ = load_b(wre16, tileId);
        cq = wmma16(ahi, bq_, cq);
        ck = wmma16(ahj, bk_, ck);
        cg = wmma16(at,  bg_, cg);
    }
    const int n = wv * 16 + (lane & 15);
    const float bqv = bq[n], bkv = bk[n], bgv = bre[n];
    float partial[8];
    #pragma unroll
    for (int v = 0; v < 8; ++v) {
        float g = silu(cg[v] + bgv);
        partial[v] = (cq[v] + bqv) * ((ck[v] + bkv) * g);
    }
    // reduce over N (16 lanes) per row
    #pragma unroll
    for (int m = 1; m < 16; m <<= 1) {
        #pragma unroll
        for (int v = 0; v < 8; ++v) partial[v] += __shfl_xor(partial[v], m, 32);
    }
    if ((lane & 15) == 0) {
        int hi8 = (lane >> 4) << 3;
        #pragma unroll
        for (int v = 0; v < 8; ++v) {
            int mrow = v + hi8;
            float a = partial[v];
            alpha[(size_t)(e0 + mrow) * HH + wv] = a;
            atomicMax(mx + is_[mrow] * HH + wv, f2u(a));
        }
    }
}

// ---------- pass 2: ex / segment sum / degree ----------
__global__ void softmax_kernel(const int* __restrict__ edge,
                               float* __restrict__ alpha,
                               const unsigned* __restrict__ mx,
                               float* __restrict__ den, int* __restrict__ deg)
{
    int idx = blockIdx.x * 256 + threadIdx.x;
    if (idx >= EE * HH) return;
    int e = idx >> 3, hh = idx & 7;
    int i_ = edge[2 * e];
    if (hh == 0) atomicAdd(deg + i_, 1);
    float m  = fmaxf(u2f(mx[i_ * HH + hh]), -1.0e9f);
    float ex = __expf(alpha[idx] - m);
    alpha[idx] = ex;
    atomAddF(den + i_ * HH + hh, ex);
}

// ---------- pass 3: v/s/tp GEMMs, coeff fuse, scatter ----------
__global__ __launch_bounds__(256) void main_kernel(
    const _Float16* __restrict__ h16, const float* __restrict__ t,
    const float* __restrict__ X1, const float* __restrict__ X2,
    const float* __restrict__ rt1, const float* __restrict__ rt2,
    const float* __restrict__ c, const int* __restrict__ edge,
    const _Float16* __restrict__ gv1w16, const float* __restrict__ gv1b,
    const _Float16* __restrict__ gv2w16, const float* __restrict__ gv2b,
    const _Float16* __restrict__ wrs16,  const float* __restrict__ brs,
    const _Float16* __restrict__ gs1w16, const float* __restrict__ gs1b,
    const _Float16* __restrict__ gs2w16, const float* __restrict__ gs2b,
    const float* __restrict__ ex, const float* __restrict__ den,
    const int* __restrict__ deg,
    float* __restrict__ h_out, float* __restrict__ X1_out, float* __restrict__ X2_out)
{
    __shared__ _Float16 hjS[16 * LSTR], tS[16 * LSTR], u1S[16 * LSTR], s1S[16 * LSTR];
    __shared__ float attnS[16][HH];
    __shared__ float rt1S[16][3], rt2S[16][5], cS[16];
    __shared__ int is_[16], js_[16];
    const int tid = threadIdx.x;
    const int e0  = blockIdx.x * 16;

    if (tid < 16) {
        is_[tid] = edge[2 * (e0 + tid)];
        js_[tid] = edge[2 * (e0 + tid) + 1];
        cS[tid]  = c[e0 + tid];
    }
    __syncthreads();

    if (tid < 128) {
        int row = tid >> 3, part = tid & 7;
        *(uint4*)(hjS + row * LSTR + part * 8) =
            ((const uint4*)(h16 + (size_t)js_[row] * DNE))[part];
    } else {
        int t2 = tid - 128, row = t2 >> 3, part = t2 & 7;
        const float* ts = t + (size_t)(e0 + row) * DNE + part * 16;
        #pragma unroll
        for (int half8 = 0; half8 < 2; ++half8) {
            float4 f0 = *(const float4*)(ts + half8 * 8);
            float4 f1 = *(const float4*)(ts + half8 * 8 + 4);
            H8 hb;
            hb.h[0] = (_Float16)f0.x; hb.h[1] = (_Float16)f0.y;
            hb.h[2] = (_Float16)f0.z; hb.h[3] = (_Float16)f0.w;
            hb.h[4] = (_Float16)f1.x; hb.h[5] = (_Float16)f1.y;
            hb.h[6] = (_Float16)f1.z; hb.h[7] = (_Float16)f1.w;
            *(uint4*)(tS + row * LSTR + part * 16 + half8 * 8) = hb.u;
        }
    }
    if (tid < 128) {
        int m = tid >> 3, hh = tid & 7;
        int i_ = is_[m];
        float exv = ex[(size_t)(e0 + m) * HH + hh];
        float d   = fmaxf(den[i_ * HH + hh], 1e-12f);
        attnS[m][hh] = exv / d * sqrtf((float)deg[i_]) * 0.08838834764831845f; // 1/sqrt(128)
    } else {
        int t2 = tid - 128, m = t2 >> 3, q = t2 & 7;
        int e = e0 + m;
        if (q < 3) rt1S[m][q]     = rt1[e * 3 + q];
        else       rt2S[m][q - 3] = rt2[e * 5 + (q - 3)];
    }
    __syncthreads();

    const int wv = tid >> 5, lane = tid & 31;
    // phase A: u1 = silu(hj@gv1w+b), s1 = silu(hj@gs1w+b) -> LDS f16
    {
        v8f aU = {}, aS = {};
        #pragma unroll
        for (int kb = 0; kb < 4; ++kb) {
            v16h a  = load_a(hjS, kb * 32);
            v16h b1 = load_b(gv1w16, kb * 8 + wv);
            v16h b2 = load_b(gs1w16, kb * 8 + wv);
            aU = wmma16(a, b1, aU);
            aS = wmma16(a, b2, aS);
        }
        const int n = wv * 16 + (lane & 15);
        const int mbase = (lane >> 4) << 3;
        const float b1v = gv1b[n], b2v = gs1b[n];
        #pragma unroll
        for (int v = 0; v < 8; ++v) {
            u1S[(mbase + v) * LSTR + n] = (_Float16)silu(aU[v] + b1v);
            s1S[(mbase + v) * LSTR + n] = (_Float16)silu(aS[v] + b2v);
        }
    }
    __syncthreads();

    // phase B: 40 column-tiles over (v, s2, tp); fuse coeff and scatter
    for (int c5 = 0; c5 < 5; ++c5) {
        const int tIdx = c5 * 8 + wv;
        const int n0   = tIdx * 16;
        const int head = n0 / 80;       // CH=80, aligned to 16
        const int seg  = n0 >> 7;       // DNE=128, aligned to 16
        const int d0   = n0 & 127;
        v8f aV = {}, aS2 = {}, aT = {};
        #pragma unroll
        for (int kb = 0; kb < 4; ++kb) {
            v16h aU  = load_a(u1S, kb * 32);
            v16h bV  = load_b(gv2w16, kb * 40 + tIdx);
            aV = wmma16(aU, bV, aV);
            v16h aSs = load_a(s1S, kb * 32);
            v16h bS  = load_b(gs2w16, kb * 40 + tIdx);
            aS2 = wmma16(aSs, bS, aS2);
            v16h aTt = load_a(tS, kb * 32);
            v16h bT  = load_b(wrs16, kb * 40 + tIdx);
            aT = wmma16(aTt, bT, aT);
        }
        const int nl = lane & 15;
        const int n  = n0 + nl;
        const int d  = d0 + nl;
        const int mbase = (lane >> 4) << 3;
        const float bv = gv2b[n], bs = gs2b[n], bt = brs[n];
        #pragma unroll
        for (int v = 0; v < 8; ++v) {
            const int m  = mbase + v;
            const int i_ = is_[m];
            float val = attnS[m][head] * (aV[v] + bv) + (aT[v] + bt) * (aS2[v] + bs) * cS[m];
            if (seg == 0) {
                atomAddF(h_out + (size_t)i_ * 128 + d, val);
            } else if (seg == 1) {
                #pragma unroll
                for (int r = 0; r < 3; ++r)
                    atomAddF(X1_out + (size_t)i_ * 384 + r * 128 + d, rt1S[m][r] * val);
            } else if (seg == 2) {
                #pragma unroll
                for (int r = 0; r < 5; ++r)
                    atomAddF(X2_out + (size_t)i_ * 640 + r * 128 + d, rt2S[m][r] * val);
            } else if (seg == 3) {
                const int j_ = js_[m];
                #pragma unroll
                for (int r = 0; r < 3; ++r)
                    atomAddF(X1_out + (size_t)i_ * 384 + r * 128 + d,
                             X1[(size_t)j_ * 384 + r * 128 + d] * val);
            } else {
                const int j_ = js_[m];
                #pragma unroll
                for (int r = 0; r < 5; ++r)
                    atomAddF(X2_out + (size_t)i_ * 640 + r * 128 + d,
                             X2[(size_t)j_ * 640 + r * 128 + d] * val);
            }
        }
    }
}

// ---------- host ----------
extern "C" void kernel_launch(void* const* d_in, const int* in_sizes, int n_in,
                              void* d_out, int out_size, void* d_ws, size_t ws_size,
                              hipStream_t stream) {
    const float* h    = (const float*)d_in[0];
    const float* t    = (const float*)d_in[1];
    const float* X1   = (const float*)d_in[2];
    const float* X2   = (const float*)d_in[3];
    const float* rt1  = (const float*)d_in[4];
    const float* rt2  = (const float*)d_in[5];
    const float* c    = (const float*)d_in[6];
    const int*   edge = (const int*)d_in[7];
    const float* Wq   = (const float*)d_in[8];  const float* bq   = (const float*)d_in[9];
    const float* Wk   = (const float*)d_in[10]; const float* bk   = (const float*)d_in[11];
    const float* Wre  = (const float*)d_in[12]; const float* bre  = (const float*)d_in[13];
    const float* gv1w = (const float*)d_in[14]; const float* gv1b = (const float*)d_in[15];
    const float* gv2w = (const float*)d_in[16]; const float* gv2b = (const float*)d_in[17];
    const float* Wrs  = (const float*)d_in[18]; const float* brs  = (const float*)d_in[19];
    const float* gs1w = (const float*)d_in[20]; const float* gs1b = (const float*)d_in[21];
    const float* gs2w = (const float*)d_in[22]; const float* gs2b = (const float*)d_in[23];

    float* out    = (float*)d_out;
    float* h_out  = out;
    float* t_out  = out + (size_t)NN * 128;
    float* X1_out = t_out + (size_t)EE * 128;
    float* X2_out = X1_out + (size_t)NN * 384;

    char* ws = (char*)d_ws;
    _Float16* wq16   = (_Float16*)(ws + 0);
    _Float16* wk16   = (_Float16*)(ws + 32768);
    _Float16* wre16  = (_Float16*)(ws + 65536);
    _Float16* gv1w16 = (_Float16*)(ws + 98304);
    _Float16* gs1w16 = (_Float16*)(ws + 131072);
    _Float16* gv2w16 = (_Float16*)(ws + 163840);
    _Float16* gs2w16 = (_Float16*)(ws + 327680);
    _Float16* wrs16  = (_Float16*)(ws + 491520);
    _Float16* h16    = (_Float16*)(ws + 655360);
    float*    alpha  = (float*)(ws + 3215360);
    unsigned* mx     = (unsigned*)(ws + 8335360);
    float*    den    = (float*)(ws + 8655360);
    int*      deg    = (int*)(ws + 8975360);

    // pass 0: pack/convert/init
    pack_b_kernel<<<64, 256, 0, stream>>>(Wq,   wq16,   128);
    pack_b_kernel<<<64, 256, 0, stream>>>(Wk,   wk16,   128);
    pack_b_kernel<<<64, 256, 0, stream>>>(Wre,  wre16,  128);
    pack_b_kernel<<<64, 256, 0, stream>>>(gv1w, gv1w16, 128);
    pack_b_kernel<<<64, 256, 0, stream>>>(gs1w, gs1w16, 128);
    pack_b_kernel<<<320, 256, 0, stream>>>(gv2w, gv2w16, 640);
    pack_b_kernel<<<320, 256, 0, stream>>>(gs2w, gs2w16, 640);
    pack_b_kernel<<<320, 256, 0, stream>>>(Wrs,  wrs16,  640);
    conv_h_kernel<<<(NN * 128 + 255) / 256, 256, 0, stream>>>(h, h16, NN * 128);
    copy_f_kernel<<<(NN * 128 + 255) / 256, 256, 0, stream>>>(h,  h_out,  NN * 128);
    copy_f_kernel<<<(NN * 384 + 255) / 256, 256, 0, stream>>>(X1, X1_out, NN * 384);
    copy_f_kernel<<<(NN * 640 + 255) / 256, 256, 0, stream>>>(X2, X2_out, NN * 640);
    hipMemcpyAsync(t_out, t, (size_t)EE * 128 * sizeof(float), hipMemcpyDeviceToDevice, stream);
    init_stats_kernel<<<(NN * HH + 255) / 256, 256, 0, stream>>>(mx, den, deg);

    // pass 1: alpha + segment max
    alpha_kernel<<<EE / 16, 256, 0, stream>>>(h16, t, edge, wq16, wk16, wre16,
                                              bq, bk, bre, alpha, mx);
    // pass 2: exp + segment sum + degree
    softmax_kernel<<<(EE * HH + 255) / 256, 256, 0, stream>>>(edge, alpha, mx, den, deg);
    // pass 3: fused v/s/tp GEMMs + scatter
    main_kernel<<<EE / 16, 256, 0, stream>>>(h16, t, X1, X2, rt1, rt2, c, edge,
                                             gv1w16, gv1b, gv2w16, gv2b, wrs16, brs,
                                             gs1w16, gs1b, gs2w16, gs2b,
                                             alpha, den, deg, h_out, X1_out, X2_out);
}
